// GCN_33767032881495
// MI455X (gfx1250) — matrix operand order repaired and checked by reading, not compiled, not run on hardware
//
#include <hip/hip_runtime.h>
#include <hip/hip_bf16.h>

typedef __attribute__((ext_vector_type(16))) _Float16 v16h;
typedef __attribute__((ext_vector_type(8)))  float    v8f;

union F16x16 { v16h v; _Float16 h[16]; };
union F32x8  { v8f  v; float    f[8];  };

#define WAVES     2
#define NTHREADS  (WAVES * 32)
#define NBLOCKS   2048
#define ALPHA_    0.2f
#define BNEPS_    1e-5f

#define DEVINL __device__ __forceinline__ __attribute__((always_inline))

// ---------------- shared memory layout (~60KB static) ----------------
// in_ : f16 input rows with circular halo, row stride 35:
//       in_[row*35 + c] = x[row][(c-5) mod 25], rows 0..24; [875,944) = zeros
//       conv1 im2col element (k,f) -> in_[k + 24*(k/11) + f]   (k = row*11+kw)
// pX/pN: pooled rows with halo, stride 16:
//       P[ci*16 + c] = p[ci][(c-2) mod 12]; tail [512,544) = zeros
//       conv2 im2col element (k,f) -> P[k + 11*(k/5) + f]      (k = ci*5+kw)
struct Smem {
  alignas(32) _Float16 w1frag[18 * 512];   // conv1 A-frags: 2 mtiles x 9 ktiles (K pad 288)
  alignas(32) _Float16 w2frag[10 * 512];   // conv2 A-frags: 2 mtiles x 5 ktiles (K=160)
  float b1[32], b2[32];
  float inv1[32], add1[32], inv2[32], add2[32];
  struct {
    _Float16 in_[944];          // haloed input branch (f16) + zero pad region
    _Float16 hA[32 * 26 + 16];  // conv output / h1f (row stride 26, +pad for spec loads)
    _Float16 hB[32 * 26 + 16];
    float    at[32 * 33];       // attention matrix (f32 for softmax)
    _Float16 g [32 * 26];       // att @ h1f
    _Float16 pX[32 * 16 + 32];  // haloed pooled layer-1 (stride 16) + zero tail
    _Float16 pN[32 * 16 + 32];
    float    x2p[32 * 6];       // pooled layer-2 outputs
    float    n2p[32 * 6];
  } ws[WAVES];
};

DEVINL v8f wmma16(v16h a, v16h b, v8f c) {
  // D = A(16x32 f16) * B(32x16 f16) + C(16x16 f32)
  return __builtin_amdgcn_wmma_f32_16x16x32_f16(false, a, false, b, (short)0, c,
                                                false, false);
}

// A-fragment element -> K index (16-bit A 16x32, ISA 7.12.2)
DEVINL int a_kidx(int e, int hh) {
  int v = e >> 1, p = e & 1;
  return ((v & 3) * 2 + p) + hh * 8 + ((v >= 4) ? 16 : 0);
}
// B-fragment element -> K index (16-bit B 32x16)
DEVINL int b_kidx(int e, int hh) {
  return a_kidx(e, hh);
}

// ---------- graph attention: a = h1 h2^T, leaky, softmax(axis=n), g = att h1 ----------
DEVINL void attention(const _Float16* h1, const _Float16* h2,
                      float* at, _Float16* g, int F, int lane) {
  const int hh  = lane >> 4;
  const int sub = lane & 15;
  const _Float16 Z = (_Float16)0.f;

  // ---- a = h1 @ h2^T : M=n1 (2 tiles), N=n2 (2 tiles), K=F pad 32 ----
  F16x16 A0, A1;
#pragma unroll
  for (int e = 0; e < 16; ++e) {
    int k = a_kidx(e, hh);
    _Float16 l0 = h1[sub * 26 + k];          // unconditional (padded buffer)
    _Float16 l1 = h1[(sub + 16) * 26 + k];
    A0.h[e] = (k < F) ? l0 : Z;
    A1.h[e] = (k < F) ? l1 : Z;
  }
  for (int nt = 0; nt < 2; ++nt) {
    int n2 = nt * 16 + sub;
    F16x16 Bf;
#pragma unroll
    for (int e = 0; e < 16; ++e) {
      int k = b_kidx(e, hh);
      _Float16 lv = h2[n2 * 26 + k];
      Bf.h[e] = (k < F) ? lv : Z;
    }
    v8f c0 = {}, c1 = {};
    c0 = wmma16(A0.v, Bf.v, c0);
    c1 = wmma16(A1.v, Bf.v, c1);
    F32x8 r0, r1; r0.v = c0; r1.v = c1;
#pragma unroll
    for (int r = 0; r < 8; ++r) {
      int n1 = r + hh * 8;
      float d0 = r0.f[r]; at[n1 * 33 + n2]        = d0 > 0.f ? d0 : ALPHA_ * d0;
      float d1 = r1.f[r]; at[(n1 + 16) * 33 + n2] = d1 > 0.f ? d1 : ALPHA_ * d1;
    }
  }
  __syncthreads();

  // ---- softmax over rows n, one column m per lane ----
  float mx = -3.0e38f;
  for (int n = 0; n < 32; ++n) mx = fmaxf(mx, at[n * 33 + lane]);
  float s = 0.f;
  for (int n = 0; n < 32; ++n) {
    float e_ = __expf(at[n * 33 + lane] - mx);
    at[n * 33 + lane] = e_;
    s += e_;
  }
  float is = 1.f / s;
  for (int n = 0; n < 32; ++n) at[n * 33 + lane] *= is;
  __syncthreads();

  // ---- g = att @ h1 : M=n (2 tiles), K=32, N=f (1 or 2 tiles) ----
  F16x16 Ag0, Ag1;
#pragma unroll
  for (int e = 0; e < 16; ++e) {
    int k = a_kidx(e, hh);
    Ag0.h[e] = (_Float16)at[sub * 33 + k];
    Ag1.h[e] = (_Float16)at[(sub + 16) * 33 + k];
  }
  const int nts = (F > 16) ? 2 : 1;
  for (int nt = 0; nt < nts; ++nt) {
    int f = nt * 16 + sub;
    F16x16 Bf;
#pragma unroll
    for (int e = 0; e < 16; ++e) {
      int k = b_kidx(e, hh);
      Bf.h[e] = h1[k * 26 + f];   // columns f >= F produce discarded D columns
    }
    v8f c0 = {}, c1 = {};
    c0 = wmma16(Ag0.v, Bf.v, c0);
    c1 = wmma16(Ag1.v, Bf.v, c1);
    F32x8 r0, r1; r0.v = c0; r1.v = c1;
    if (f < F) {
#pragma unroll
      for (int r = 0; r < 8; ++r) {
        int n = r + hh * 8;
        g[n * 26 + f]        = (_Float16)r0.f[r];
        g[(n + 16) * 26 + f] = (_Float16)r1.f[r];
      }
    }
  }
  __syncthreads();
}

// ---------- conv2: out[o,f] = b2[o] + sum_{ci,kw} P[ci,(f+kw-2)%12] w2[o,ci,kw] ----------
DEVINL void conv2run(const _Float16* P, _Float16* H, const Smem& sm, int lane) {
  const int hh  = lane >> 4;
  const int sub = lane & 15;
  const int f   = sub;                  // single N tile (12 <= 16)
  v8f acc0 = {}, acc1 = {};
  for (int kt = 0; kt < 5; ++kt) {
    F16x16 Bf;
#pragma unroll
    for (int e = 0; e < 16; ++e) {
      unsigned k = kt * 32 + b_kidx(e, hh);        // k = ci*5 + kw
      unsigned idx = k + 11u * (k / 5u) + f;       // = ci*16 + kw + f
      Bf.h[e] = P[idx];
    }
    const v16h a0 = *(const v16h*)&sm.w2frag[kt * 512 + lane * 16];
    const v16h a1 = *(const v16h*)&sm.w2frag[(5 + kt) * 512 + lane * 16];
    acc0 = wmma16(a0, Bf.v, acc0);
    acc1 = wmma16(a1, Bf.v, acc1);
  }
  F32x8 r0, r1; r0.v = acc0; r1.v = acc1;
  if (f < 12) {
#pragma unroll
    for (int r = 0; r < 8; ++r) {
      int o = r + hh * 8;
      H[o * 26 + f]        = (_Float16)(r0.f[r] + sm.b2[o]);
      H[(o + 16) * 26 + f] = (_Float16)(r1.f[r] + sm.b2[o + 16]);
    }
  }
}

__global__ __launch_bounds__(NTHREADS)
void gcn_fused(const float* __restrict__ x,
               const float* __restrict__ w1, const float* __restrict__ b1g,
               const float* __restrict__ g1, const float* __restrict__ be1,
               const float* __restrict__ mu1, const float* __restrict__ va1,
               const float* __restrict__ w2, const float* __restrict__ b2g,
               const float* __restrict__ g2, const float* __restrict__ be2,
               const float* __restrict__ mu2, const float* __restrict__ va2,
               const float* __restrict__ lw, const float* __restrict__ lb,
               float* __restrict__ out, int nItems) {
  __shared__ Smem sm;
  const int t = threadIdx.x;

  // ---------------- per-block setup ----------------
  for (int i = t; i < 18 * 512; i += NTHREADS) {      // conv1 weight fragments
    int fid = i >> 9, r = i & 511, ln = r >> 4, e = r & 15;
    int mt = fid / 9, kt = fid % 9;
    int o  = (ln & 15) + mt * 16;
    int k  = kt * 32 + a_kidx(e, ln >> 4);
    sm.w1frag[i] = (k < 275) ? (_Float16)w1[o * 275 + k] : (_Float16)0.f;
  }
  for (int i = t; i < 10 * 512; i += NTHREADS) {      // conv2 weight fragments
    int fid = i >> 9, r = i & 511, ln = r >> 4, e = r & 15;
    int mt = fid / 5, kt = fid % 5;
    int o  = (ln & 15) + mt * 16;
    int k  = kt * 32 + a_kidx(e, ln >> 4);
    sm.w2frag[i] = (k < 160) ? (_Float16)w2[o * 160 + k] : (_Float16)0.f;
  }
  if (t < 32) {
    sm.b1[t] = b1g[t];
    sm.b2[t] = b2g[t];
    float iv1 = g1[t] / sqrtf(va1[t] + BNEPS_);
    sm.inv1[t] = iv1; sm.add1[t] = be1[t] - mu1[t] * iv1;
    float iv2 = g2[t] / sqrtf(va2[t] + BNEPS_);
    sm.inv2[t] = iv2; sm.add2[t] = be2[t] - mu2[t] * iv2;
  }

  const int wave = t >> 5, lane = t & 31;
  const int hh = lane >> 4, sub = lane & 15;
  auto& W = sm.ws[wave];

  // zero pad regions once (never overwritten afterwards)
  for (int i = 875 + lane; i < 944; i += 32) W.in_[i] = (_Float16)0.f;
  for (int i = 512 + lane; i < 544; i += 32) {
    W.pX[i] = (_Float16)0.f;
    W.pN[i] = (_Float16)0.f;
  }
  __syncthreads();

  const int gw = blockIdx.x * WAVES + wave;
  const int nw = NBLOCKS * WAVES;
  const int iters = (nItems + nw - 1) / nw;

  for (int it = 0; it < iters; ++it) {
    int item = gw + it * nw;
    bool active = item < nItems;
    const float* xi = x + (size_t)(active ? item : 0) * 1250;

    // ================= layer 1 : two branches (xa, na) =================
    for (int br = 0; br < 2; ++br) {
      // staged load with circular halo: in_[row*35 + c] = x[row][(c-5) mod 25]
      for (int i = lane; i < 875; i += 32) {
        int row = i / 35, c = i % 35;
        int w = (c + 20) % 25;          // (c - 5 + 25) % 25
        W.in_[i] = (_Float16)xi[row * 50 + br * 25 + w];
      }
      __syncthreads();

      // conv1: D[o,f] = sum_k w1[o,k] * xa[c,kh,(f+kw-5)%25]
      // kt outer, 4 accumulators: A-frags loaded once, B columns share addresses
      {
        v8f a00 = {}, a10 = {}, a01 = {}, a11 = {};
        for (int kt = 0; kt < 9; ++kt) {
          F16x16 B0, B1;
#pragma unroll
          for (int e = 0; e < 16; ++e) {
            unsigned k = kt * 32 + b_kidx(e, hh);        // k = row*11 + kw
            unsigned idx = k + 24u * (k / 11u) + sub;    // = row*35 + kw + f0
            B0.h[e] = W.in_[idx];                        // f0 = sub
            B1.h[e] = W.in_[idx + 16];                   // f1 = sub + 16
          }
          const v16h a0 = *(const v16h*)&sm.w1frag[kt * 512 + lane * 16];
          const v16h a1 = *(const v16h*)&sm.w1frag[(9 + kt) * 512 + lane * 16];
          a00 = wmma16(a0, B0.v, a00);
          a10 = wmma16(a1, B0.v, a10);
          a01 = wmma16(a0, B1.v, a01);
          a11 = wmma16(a1, B1.v, a11);
        }
        F32x8 r00, r10, r01, r11;
        r00.v = a00; r10.v = a10; r01.v = a01; r11.v = a11;
        const int f0 = sub;            // always < 25
#pragma unroll
        for (int r = 0; r < 8; ++r) {
          int o = r + hh * 8;
          W.hA[o * 26 + f0]        = (_Float16)(r00.f[r] + sm.b1[o]);
          W.hA[(o + 16) * 26 + f0] = (_Float16)(r10.f[r] + sm.b1[o + 16]);
        }
        if (sub < 9) {                 // f1 = sub+16 < 25
          const int f1 = sub + 16;
#pragma unroll
          for (int r = 0; r < 8; ++r) {
            int o = r + hh * 8;
            W.hA[o * 26 + f1]        = (_Float16)(r01.f[r] + sm.b1[o]);
            W.hA[(o + 16) * 26 + f1] = (_Float16)(r11.f[r] + sm.b1[o + 16]);
          }
        }
      }
      __syncthreads();

      attention(W.hA, W.hA, W.at, W.g, 25, lane);   // h1 == h2 in layer 1

      // bn1 + pool(25->12) + relu, store with circular halo (stride 16)
      {
        _Float16* P = (br == 0) ? W.pX : W.pN;
        float iv = sm.inv1[lane], ad = sm.add1[lane];
        int base = lane * 16;
        _Float16 v0h, v1h, v10h, v11h;
        for (int f = 0; f < 12; ++f) {
          float u0 = (float)W.g[lane * 26 + 2 * f]     * iv + ad;
          float u1 = (float)W.g[lane * 26 + 2 * f + 1] * iv + ad;
          _Float16 pv = (_Float16)fmaxf(fmaxf(u0, u1), 0.f);
          P[base + 2 + f] = pv;
          if (f == 0)  v0h  = pv;
          if (f == 1)  v1h  = pv;
          if (f == 10) v10h = pv;
          if (f == 11) v11h = pv;
        }
        P[base + 0]  = v10h;   // (0-2) mod 12 = 10
        P[base + 1]  = v11h;
        P[base + 14] = v0h;    // (14-2) mod 12 = 0
        P[base + 15] = v1h;
      }
      __syncthreads();
    }

    // ================= layer 2 =================
    conv2run(W.pX, W.hA, sm, lane);
    conv2run(W.pN, W.hB, sm, lane);
    __syncthreads();

    // x2 = graph_conv(xa_p, na_p): h1 = conv(xa_p), h2 = conv(na_p)
    attention(W.hA, W.hB, W.at, W.g, 12, lane);
    {
      float iv = sm.inv2[lane], ad = sm.add2[lane];
      for (int f = 0; f < 6; ++f) {
        float u0 = (float)W.g[lane * 26 + 2 * f]     * iv + ad;
        float u1 = (float)W.g[lane * 26 + 2 * f + 1] * iv + ad;
        W.x2p[lane * 6 + f] = fmaxf(fmaxf(u0, u1), 0.f);
      }
    }
    __syncthreads();

    // n2 = graph_conv(na_p, xa_p): h1 = conv(na_p), h2 = conv(xa_p)
    attention(W.hB, W.hA, W.at, W.g, 12, lane);
    {
      float iv = sm.inv2[lane], ad = sm.add2[lane];
      for (int f = 0; f < 6; ++f) {
        float u0 = (float)W.g[lane * 26 + 2 * f]     * iv + ad;
        float u1 = (float)W.g[lane * 26 + 2 * f + 1] * iv + ad;
        W.n2p[lane * 6 + f] = fmaxf(fmaxf(u0, u1), 0.f);
      }
    }
    __syncthreads();

    // ----- final linear: flat(384) @ lin_w^T + lin_b -----
    if (active && lane < 13) {
      float s = lb[lane];
      const float* wr = lw + lane * 384;
      for (int n = 0; n < 32; ++n) {
#pragma unroll
        for (int f = 0; f < 12; ++f) {
          float xv = (f < 6) ? W.x2p[n * 6 + f] : W.n2p[n * 6 + f - 6];
          s += xv * wr[n * 12 + f];
        }
      }
      out[(size_t)item * 13 + lane] = s;
    }
    __syncthreads();
  }
}

extern "C" void kernel_launch(void* const* d_in, const int* in_sizes, int n_in,
                              void* d_out, int out_size, void* d_ws, size_t ws_size,
                              hipStream_t stream) {
  (void)n_in; (void)out_size; (void)d_ws; (void)ws_size;
  const float* x   = (const float*)d_in[0];
  const float* w1  = (const float*)d_in[1];
  const float* b1  = (const float*)d_in[2];
  const float* g1  = (const float*)d_in[3];
  const float* be1 = (const float*)d_in[4];
  const float* mu1 = (const float*)d_in[5];
  const float* va1 = (const float*)d_in[6];
  const float* w2  = (const float*)d_in[7];
  const float* b2  = (const float*)d_in[8];
  const float* g2  = (const float*)d_in[9];
  const float* be2 = (const float*)d_in[10];
  const float* mu2 = (const float*)d_in[11];
  const float* va2 = (const float*)d_in[12];
  const float* lw  = (const float*)d_in[13];
  const float* lb  = (const float*)d_in[14];
  float* out = (float*)d_out;
  int nItems = in_sizes[0] / 1250;   // (B, 5, 5, 50)

  gcn_fused<<<NBLOCKS, NTHREADS, 0, stream>>>(
      x, w1, b1, g1, be1, mu1, va1, w2, b2, g2, be2, mu2, va2, lw, lb,
      out, nItems);
}
